// Attention_919123001749
// MI455X (gfx1250) — compile-verified, matrix operand order checked by reading, not run on hardware
//
#include <hip/hip_runtime.h>
#include <hip/hip_bf16.h>

// ---------------------------------------------------------------------------
// Attention pipeline for MI455X (gfx1250, wave32, WMMA bf16 16x16x32)
//   keys    = tanh(tanh(x @ Kw1^T) @ Kw2^T)
//   queries = tanh(tanh(x @ Qw1^T) @ Qw2^T)
//   P       = softmax(keys @ queries^T / 32)
//   out     = P @ x        (written as [S, B, D])
// GEMMs: bf16 WMMA (v_wmma_f32_16x16x32_bf16), f32 accumulation.
// Staging: global_load_async_to_lds_b128 (ASYNCcnt) + double-buffered LDS.
// ---------------------------------------------------------------------------

typedef __bf16 bf16;
typedef __bf16 v8bf  __attribute__((ext_vector_type(8)));
typedef __bf16 v16bf __attribute__((ext_vector_type(16)));
typedef float  v8f   __attribute__((ext_vector_type(8)));

#define S_DIM 2048
#define B_DIM 8
#define D_DIM 1024

// Block tile 256x128, K-step 32. 8 waves in a 4(M) x 2(N) grid,
// each wave computes a 64x64 tile = 4x4 WMMA accumulators.
#define BM 256
#define BN 128
#define BK 32
#define LDS_K 40   // padded row length in halfs (80B rows -> conflict-free)

// ---------------------------------------------------------------------------
// CDNA5 async DMA: global -> LDS, 16B per lane, tracked on ASYNCcnt.
// lds_off = LDS byte offset (low 32 bits of the generic shared pointer).
// ---------------------------------------------------------------------------
__device__ __forceinline__ void async_load_b128(unsigned lds_off,
                                                const void* gaddr) {
    asm volatile("global_load_async_to_lds_b128 %0, %1, off"
                 :: "v"(lds_off), "v"(gaddr)
                 : "memory");
}
__device__ __forceinline__ void wait_async0() {
    asm volatile("s_wait_asynccnt 0x0" ::: "memory");
}
__device__ __forceinline__ unsigned lds_off(const void* p) {
    return (unsigned)(unsigned long long)p;
}

// ---------------------------------------------------------------------------
// Fragment load from an LDS row (K-major). CDNA5 16-bit A/B operand layout:
// lane L: m/n = L%16, h = L/16; frag halves [0..7]  = K[8h .. 8h+7]
//                               frag halves [8..15] = K[16+8h .. 16+8h+7]
// ---------------------------------------------------------------------------
__device__ __forceinline__ v16bf load_frag(const bf16* row, int h) {
    v8bf lo = *(const v8bf*)(row + h * 8);
    v8bf hi = *(const v8bf*)(row + 16 + h * 8);
    return __builtin_shufflevector(lo, hi, 0, 1, 2, 3, 4, 5, 6, 7,
                                           8, 9, 10, 11, 12, 13, 14, 15);
}

// ---------------------------------------------------------------------------
// Batched bf16 GEMM:  C[z][m][n] = epi( sum_k A[z][m][k] * B[z][n][k] )
// Both operands are K-major with leading dimension K (lda == ldb == K).
// EPI 0: C is bf16, epi = tanh(acc)
// EPI 1: C is f32,  epi = acc * alpha
// ---------------------------------------------------------------------------
template <int EPI>
__global__ __launch_bounds__(256)
void gemm_bf16(const bf16* __restrict__ A, const bf16* __restrict__ B,
               void* __restrict__ Cv, int K, int ldc,
               long strideA, long strideB, long strideC, float alpha) {
    __shared__ __align__(16) bf16 As[2][BM * LDS_K];   // 2 x 20 KB
    __shared__ __align__(16) bf16 Bs[2][BN * LDS_K];   // 2 x 10 KB

    const int tid  = threadIdx.x;
    const int wave = tid >> 5;          // 0..7
    const int lane = tid & 31;
    const int wm   = wave >> 1;         // 0..3  (M strip of 64)
    const int wn   = wave & 1;          // 0..1  (N strip of 64)
    const int l16  = lane & 15;
    const int h    = lane >> 4;

    const bf16* Ab = A + strideA * blockIdx.z + (long)(blockIdx.y * BM) * K;
    const bf16* Bb = B + strideB * blockIdx.z + (long)(blockIdx.x * BN) * K;

    // Per-thread staging chunks (16B = 8 halfs each):
    // A tile: 256x32 halfs = 1024 chunks -> 4/thread
    // B tile: 128x32 halfs =  512 chunks -> 2/thread
    int arow[4], akc[4], brow[2], bkc[2];
#pragma unroll
    for (int i = 0; i < 4; ++i) {
        const int c = tid + i * 256;
        arow[i] = c >> 2;
        akc[i]  = (c & 3) * 8;
    }
#pragma unroll
    for (int i = 0; i < 2; ++i) {
        const int c = tid + i * 256;
        brow[i] = c >> 2;
        bkc[i]  = (c & 3) * 8;
    }

    v8f acc[4][4] = {};

    const int nk = K / BK;

    // Prologue: stage slab 0 into buffer 0.
#pragma unroll
    for (int i = 0; i < 4; ++i)
        async_load_b128(lds_off(&As[0][arow[i] * LDS_K + akc[i]]),
                        &Ab[(long)arow[i] * K + akc[i]]);
#pragma unroll
    for (int i = 0; i < 2; ++i)
        async_load_b128(lds_off(&Bs[0][brow[i] * LDS_K + bkc[i]]),
                        &Bb[(long)brow[i] * K + bkc[i]]);

    for (int ik = 0; ik < nk; ++ik) {
        const int cur = ik & 1;
        wait_async0();          // current buffer's DMA complete (this wave)
        __syncthreads();        // ... and every wave's

        if (ik + 1 < nk) {      // stream next slab into the other buffer
            const int nxt = cur ^ 1;
            const long k1 = (long)(ik + 1) * BK;
#pragma unroll
            for (int i = 0; i < 4; ++i)
                async_load_b128(lds_off(&As[nxt][arow[i] * LDS_K + akc[i]]),
                                &Ab[(long)arow[i] * K + k1 + akc[i]]);
#pragma unroll
            for (int i = 0; i < 2; ++i)
                async_load_b128(lds_off(&Bs[nxt][brow[i] * LDS_K + bkc[i]]),
                                &Bb[(long)brow[i] * K + k1 + bkc[i]]);
        }

        v16bf af[4], bfv[4];
#pragma unroll
        for (int mt = 0; mt < 4; ++mt)
            af[mt] = load_frag(&As[cur][(wm * 64 + mt * 16 + l16) * LDS_K], h);
#pragma unroll
        for (int nt = 0; nt < 4; ++nt)
            bfv[nt] = load_frag(&Bs[cur][(wn * 64 + nt * 16 + l16) * LDS_K], h);

#pragma unroll
        for (int mt = 0; mt < 4; ++mt)
#pragma unroll
            for (int nt = 0; nt < 4; ++nt)
                acc[mt][nt] = __builtin_amdgcn_wmma_f32_16x16x32_bf16(
                    false, af[mt], false, bfv[nt], (short)0, acc[mt][nt],
                    false, false);
        // No trailing barrier needed: buffer `cur` is only overwritten after
        // the next iteration's top barrier, which follows all reads above.
    }

    // -- epilogue: lane L holds n = L%16, rows m = 8*(L/16) + r (r = VGPR) --
    const int gn0 = blockIdx.x * BN + wn * 64;
    const int gm0 = blockIdx.y * BM + wm * 64 + 8 * h;
#pragma unroll
    for (int mt = 0; mt < 4; ++mt) {
#pragma unroll
        for (int nt = 0; nt < 4; ++nt) {
#pragma unroll
            for (int r = 0; r < 8; ++r) {
                const long gm = gm0 + mt * 16 + r;
                const long gn = gn0 + nt * 16 + l16;
                const float v = acc[mt][nt][r];
                if (EPI == 0) {
                    bf16* C = (bf16*)Cv + strideC * blockIdx.z;
                    C[gm * ldc + gn] = (bf16)tanhf(v);
                } else {
                    float* C = (float*)Cv + strideC * blockIdx.z;
                    C[gm * ldc + gn] = v * alpha;
                }
            }
        }
    }
}

// ---------------------------------------------------------------------------
// sequence [S][B][D] f32  ->  xf [B*S][D] bf16  and  xT [B][D][S] bf16
// ---------------------------------------------------------------------------
__global__ __launch_bounds__(256)
void convert_x(const float* __restrict__ seq, bf16* __restrict__ xf,
               bf16* __restrict__ xT) {
    const long idx = (long)blockIdx.x * 256 + threadIdx.x;
    const int  d  = idx & (D_DIM - 1);
    const long sb = idx >> 10;
    const int  b  = sb & (B_DIM - 1);
    const long s  = sb >> 3;
    const float v = seq[idx];
    xf[((long)b * S_DIM + s) * D_DIM + d] = (bf16)v;
    xT[(long)b * D_DIM * S_DIM + (long)d * S_DIM + s] = (bf16)v;
}

__global__ __launch_bounds__(256)
void convert_f32_bf16(const float* __restrict__ in, bf16* __restrict__ out) {
    const long i = (long)blockIdx.x * 256 + threadIdx.x;
    out[i] = (bf16)in[i];
}

// ---------------------------------------------------------------------------
// Row softmax over 2048 f32 -> bf16   (one block per row, 8 elems/thread)
// ---------------------------------------------------------------------------
__global__ __launch_bounds__(256)
void softmax_rows(const float* __restrict__ S, bf16* __restrict__ P) {
    const long row = blockIdx.x;
    const float* s = S + row * S_DIM;
    bf16*        p = P + row * S_DIM;
    __shared__ float red[256];
    const int tid = threadIdx.x;

    float m = -3.4e38f;
#pragma unroll
    for (int i = 0; i < 8; ++i) m = fmaxf(m, s[tid + i * 256]);
    red[tid] = m;
    __syncthreads();
    for (int off = 128; off > 0; off >>= 1) {
        if (tid < off) red[tid] = fmaxf(red[tid], red[tid + off]);
        __syncthreads();
    }
    m = red[0];
    __syncthreads();

    float e[8], sum = 0.f;
#pragma unroll
    for (int i = 0; i < 8; ++i) {
        e[i] = __expf(s[tid + i * 256] - m);
        sum += e[i];
    }
    red[tid] = sum;
    __syncthreads();
    for (int off = 128; off > 0; off >>= 1) {
        if (tid < off) red[tid] += red[tid + off];
        __syncthreads();
    }
    const float inv = 1.0f / red[0];
#pragma unroll
    for (int i = 0; i < 8; ++i) p[tid + i * 256] = (bf16)(e[i] * inv);
}

// ---------------------------------------------------------------------------
extern "C" void kernel_launch(void* const* d_in, const int* in_sizes, int n_in,
                              void* d_out, int out_size, void* d_ws,
                              size_t ws_size, hipStream_t stream) {
    const float* seq = (const float*)d_in[0];
    const float* Kw1 = (const float*)d_in[1];
    const float* Kw2 = (const float*)d_in[2];
    const float* Qw1 = (const float*)d_in[3];
    const float* Qw2 = (const float*)d_in[4];

    char* ws = (char*)d_ws;
    const long MB = 1ll << 20;
    bf16*  xf      = (bf16*)(ws);                 //  32 MB [B*S][D]
    bf16*  xT      = (bf16*)(ws + 32 * MB);       //  32 MB [B][D][S]
    bf16*  Wk1     = (bf16*)(ws + 64 * MB);       //   8 MB (4 x 2 MB)
    bf16*  Wk2     = Wk1 + (long)D_DIM * D_DIM;
    bf16*  Wq1     = Wk2 + (long)D_DIM * D_DIM;
    bf16*  Wq2     = Wq1 + (long)D_DIM * D_DIM;
    bf16*  hbuf    = (bf16*)(ws + 72 * MB);       //  32 MB [B*S][D]
    bf16*  keys    = (bf16*)(ws + 104 * MB);      //  32 MB
    bf16*  queries = (bf16*)(ws + 136 * MB);      //  32 MB
    float* scores  = (float*)(ws + 168 * MB);     // 128 MB [B][S][S]
    bf16*  P       = (bf16*)(ws + 296 * MB);      //  64 MB [B][S][S]
    float* out     = (float*)d_out;               //  [S][B][D]

    const dim3 blk(256);

    // --- precision conversion ---
    convert_x<<<dim3(65536), blk, 0, stream>>>(seq, xf, xT);
    convert_f32_bf16<<<dim3(4096), blk, 0, stream>>>(Kw1, Wk1);
    convert_f32_bf16<<<dim3(4096), blk, 0, stream>>>(Kw2, Wk2);
    convert_f32_bf16<<<dim3(4096), blk, 0, stream>>>(Qw1, Wq1);
    convert_f32_bf16<<<dim3(4096), blk, 0, stream>>>(Qw2, Wq2);

    // --- MLPs: M = B*S = 16384, N = K = 1024 ---
    const dim3 gMLP(D_DIM / BN, (B_DIM * S_DIM) / BM, 1);
    gemm_bf16<0><<<gMLP, blk, 0, stream>>>(xf,   Wk1, hbuf,    D_DIM, D_DIM, 0, 0, 0, 0.f);
    gemm_bf16<0><<<gMLP, blk, 0, stream>>>(hbuf, Wk2, keys,    D_DIM, D_DIM, 0, 0, 0, 0.f);
    gemm_bf16<0><<<gMLP, blk, 0, stream>>>(xf,   Wq1, hbuf,    D_DIM, D_DIM, 0, 0, 0, 0.f);
    gemm_bf16<0><<<gMLP, blk, 0, stream>>>(hbuf, Wq2, queries, D_DIM, D_DIM, 0, 0, 0, 0.f);

    // --- scores[b] = keys_b @ queries_b^T / sqrt(D) ---
    const dim3 gSc(S_DIM / BN, S_DIM / BM, B_DIM);
    gemm_bf16<1><<<gSc, blk, 0, stream>>>(keys, queries, scores, D_DIM, S_DIM,
                                          (long)S_DIM * D_DIM,
                                          (long)S_DIM * D_DIM,
                                          (long)S_DIM * S_DIM, 0.03125f);

    // --- softmax rows -> bf16 P ---
    softmax_rows<<<dim3(B_DIM * S_DIM), blk, 0, stream>>>(scores, P);

    // --- attended: out[s][b][d] = sum_t P[b][s][t] * x[b][t][d] ---
    const dim3 gAt(D_DIM / BN, S_DIM / BM, B_DIM);
    gemm_bf16<1><<<gAt, blk, 0, stream>>>(P, xT, out, S_DIM, B_DIM * D_DIM,
                                          (long)S_DIM * S_DIM,
                                          (long)D_DIM * S_DIM,
                                          (long)D_DIM, 1.0f);
}
